// PerformerAttentionCore_35923106463823
// MI455X (gfx1250) — matrix operand order repaired and checked by reading, not compiled, not run on hardware
//
#include <hip/hip_runtime.h>
#include <math.h>
#include <stdint.h>

typedef __attribute__((ext_vector_type(16))) _Float16 v16h;
typedef __attribute__((ext_vector_type(8)))  float    v8f;
typedef __attribute__((ext_vector_type(4)))  int      v4i;

#define B_  4
#define H_  16
#define N_  4096
#define D_  64
#define DH_ 32                      // D-half handled per block
#define M_  128
#define C_  32                      // time-chunk
#define NCHUNK (N_ / C_)            // 128
#define SCALE      0.35355339059327373f  // 64^-0.25
#define INV_SQRT_M 0.08838834764831845f  // 1/sqrt(128)
#define EPS_PHI 1e-4f
#define EPS_DEN 1e-6f

// ---------------- async global->LDS copy (CDNA5 ASYNCcnt path) --------------
#if __has_builtin(__builtin_amdgcn_global_load_async_to_lds_b128) && \
    __has_builtin(__builtin_amdgcn_s_wait_asynccnt)
#define HAS_ASYNC_LDS 1
typedef __attribute__((address_space(1))) v4i* gas_v4i;
typedef __attribute__((address_space(3))) v4i* lds_v4i;
__device__ __forceinline__ void cp_async16(void* lds, const void* g) {
  // AS1 value == flat value for global; AS3 value == low 32 bits of flat (ISA 10.2)
  __builtin_amdgcn_global_load_async_to_lds_b128(
      (gas_v4i)(uintptr_t)g, (lds_v4i)(uint32_t)(uintptr_t)lds, 0, 0);
}
__device__ __forceinline__ void wait_async_all() {
  __builtin_amdgcn_s_wait_asynccnt(0);
}
#else
#define HAS_ASYNC_LDS 0
__device__ __forceinline__ void cp_async16(void* lds, const void* g) {
  *(float4*)lds = *(const float4*)g;       // synchronous fallback
}
__device__ __forceinline__ void wait_async_all() {}
#endif

// ---------------- WMMA fragment load/store (LDS row-major sources) ----------
// A/B loaders read two 8-wide k runs per lane -> compiler emits ds_load_b128.
__device__ __forceinline__ v16h load_a(const _Float16* p, int ld, int r0, int k0) {
  int l = threadIdx.x & 31, m = l & 15, s = l >> 4;
  v16h a;
#pragma unroll
  for (int j = 0; j < 16; ++j) {
    int k = ((j >> 3) << 4) + (s << 3) + (j & 7);
    a[j] = p[(r0 + m) * ld + k0 + k];
  }
  return a;
}
// B (32x16) from TRANSPOSED source pT[n][k]
__device__ __forceinline__ v16h load_bT(const _Float16* p, int ld, int k0, int c0) {
  int l = threadIdx.x & 31, n = l & 15, s = l >> 4;
  v16h b;
#pragma unroll
  for (int j = 0; j < 16; ++j) {
    int k = ((j >> 3) << 4) + (s << 3) + (j & 7);
    b[j] = p[(c0 + n) * ld + k0 + k];
  }
  return b;
}
__device__ __forceinline__ v8f load_c(const float* p, int ld, int r0, int c0) {
  int l = threadIdx.x & 31, n = l & 15, s = l >> 4;
  v8f c;
#pragma unroll
  for (int j = 0; j < 8; ++j) c[j] = p[(r0 + j + 8 * s) * ld + c0 + n];
  return c;
}
__device__ __forceinline__ void store_c(float* p, int ld, int r0, int c0, v8f c) {
  int l = threadIdx.x & 31, n = l & 15, s = l >> 4;
#pragma unroll
  for (int j = 0; j < 8; ++j) p[(r0 + j + 8 * s) * ld + c0 + n] = c[j];
}
// store f32 tile + f16 TRANSPOSED shadow (ShT[col][row], row length M_)
__device__ __forceinline__ void store_c_dualT(float* pf, _Float16* phT, int ld,
                                              int r0, int c0, v8f c) {
  int l = threadIdx.x & 31, n = l & 15, s = l >> 4;
#pragma unroll
  for (int j = 0; j < 8; ++j) {
    int r = r0 + j + 8 * s;
    float v = c[j];
    pf[r * ld + c0 + n] = v;
    phT[(c0 + n) * M_ + r] = (_Float16)v;
  }
}

#define WMMA_F16(a, b, c) \
  __builtin_amdgcn_wmma_f32_16x16x32_f16(false, (a), false, (b), (short)0, (c), false, false)

// proj[32][128] = x[32][64] @ omega^T (omega row-major [128][64] == B^T layout)
template <int NWAVES>
__device__ __forceinline__ void proj_wmma(const _Float16* x, const _Float16* om,
                                          float* proj) {
  int wave = (int)(threadIdx.x >> 5);
  for (int t = wave; t < 16; t += NWAVES) {
    int rt = t >> 3, ct = t & 7;
    v8f acc = {};
#pragma unroll
    for (int k0 = 0; k0 < 64; k0 += 32) {
      v16h a = load_a(x, 64, rt * 16, k0);
      v16h b = load_bT(om, 64, k0, ct * 16);
      acc = WMMA_F16(a, b, acc);
    }
    store_c(proj, 128, rt * 16, ct * 16, acc);
  }
}

__device__ __forceinline__ unsigned enc_f32(float f) {
  unsigned u = __float_as_uint(f);
  return (u & 0x80000000u) ? ~u : (u | 0x80000000u);
}
__device__ __forceinline__ float dec_f32(unsigned u) {
  return (u & 0x80000000u) ? __uint_as_float(u ^ 0x80000000u) : __uint_as_float(~u);
}

// =================== Pass 1: global max of (k*scale)Ω^T - 0.5||k*scale||^2 ===
__global__ __launch_bounds__(256) void kmax_kernel(const float* __restrict__ kg,
                                                   const float* __restrict__ omg,
                                                   unsigned* __restrict__ ws) {
  __shared__ _Float16 omh[M_ * D_];
  __shared__ _Float16 xk[C_ * D_];
  __shared__ float proj[C_ * M_];
  __shared__ float sqk[C_];
  __shared__ float red[256];

  const int tid = threadIdx.x;
  const int chunk = blockIdx.x % NCHUNK;
  const int bh = blockIdx.x / NCHUNK;
  const float* kp = kg + ((size_t)bh * N_ + (size_t)chunk * C_) * D_;

  for (int e = tid; e < M_ * D_; e += 256) omh[e] = (_Float16)omg[e];
  if (tid < C_) sqk[tid] = 0.0f;
  __syncthreads();
  for (int e = tid; e < C_ * D_; e += 256) {
    float f = kp[e] * SCALE;
    xk[e] = (_Float16)f;
    atomicAdd(&sqk[e >> 6], 0.5f * f * f);
  }
  __syncthreads();

  proj_wmma<8>(xk, omh, proj);
  __syncthreads();

  float lmax = -3.0e38f;
  for (int e = tid; e < C_ * M_; e += 256)
    lmax = fmaxf(lmax, proj[e] - sqk[e >> 7]);
  red[tid] = lmax;
  __syncthreads();
  for (int s = 128; s > 0; s >>= 1) {
    if (tid < s) red[tid] = fmaxf(red[tid], red[tid + s]);
    __syncthreads();
  }
  if (tid == 0) atomicMax(ws, enc_f32(red[0]));
}

// =================== Pass 2: chunked causal scan, one (b,h,D-half)/block =====
__global__ __launch_bounds__(512) void performer_kernel(const float* __restrict__ qg,
                                                        const float* __restrict__ kg,
                                                        const float* __restrict__ vg,
                                                        const float* __restrict__ omg,
                                                        const unsigned* __restrict__ ws,
                                                        float* __restrict__ outg) {
  __shared__ float     S[M_ * DH_];       // 16 KB  running state (f32)
  __shared__ _Float16  ShT[DH_ * M_];     //  8 KB  f16 transposed shadow [d][m]
  __shared__ _Float16  omh[M_ * D_];      // 16 KB
  __shared__ _Float16  xq[C_ * D_];       //  4 KB
  __shared__ _Float16  xk[C_ * D_];       //  4 KB
  __shared__ _Float16  xvT[DH_ * C_];     //  2 KB  v chunk transposed [d][i]
  __shared__ float     proj[C_ * M_];     // 16 KB
  __shared__ _Float16  pq[C_ * M_];       //  8 KB  phi_q  [i][m]
  __shared__ _Float16  pk[C_ * M_];       //  8 KB  phi_k  [i][m]
  __shared__ _Float16  pkT[M_ * C_];      //  8 KB  phi_k transposed [m][i]
  __shared__ float     o[C_ * DH_];       //  4 KB  output accumulator
  __shared__ float     sa[C_ * C_];       //  4 KB  A = pq pk^T
  __shared__ _Float16  am[C_ * C_];       //  2 KB  causal-masked A (f16)
  __shared__ __align__(16) float qs[2 * C_ * D_];   // 16 KB double-buffered stage
  __shared__ __align__(16) float ks[2 * C_ * D_];   // 16 KB
  __shared__ __align__(16) float vs[2 * C_ * DH_];  //  8 KB
  __shared__ float     z[M_];
  __shared__ float     sqk[C_], rmax[C_], zq[C_], den[C_];
  __shared__ float     s_gmax;

  const int tid = threadIdx.x;
  const int wave = tid >> 5;
  const int bh = blockIdx.x >> 1;
  const int dco = (blockIdx.x & 1) * DH_;   // column offset of this D-half

  const float* qb = qg + (size_t)bh * N_ * D_;
  const float* kb = kg + (size_t)bh * N_ * D_;
  const float* vb = vg + (size_t)bh * N_ * D_;
  float* ob = outg + (size_t)bh * N_ * D_;

  // ---- init ----
  for (int e = tid; e < M_ * D_; e += 512) omh[e] = (_Float16)omg[e];
  for (int e = tid; e < M_ * DH_; e += 512) {
    S[e] = 0.0f;
    ShT[e] = (_Float16)0.0f;
  }
  if (tid < M_) z[tid] = 0.0f;
  if (tid == 0) s_gmax = dec_f32(ws[0]);

  // ---- prologue: issue async copies for chunk 0 into buffer 0 ----
  cp_async16(&qs[tid * 4], qb + tid * 4);
  cp_async16(&ks[tid * 4], kb + tid * 4);
  if (tid < 256) {
    int i = tid >> 3, seg = tid & 7;
    cp_async16(&vs[i * DH_ + seg * 4], vb + i * D_ + dco + seg * 4);
  }
  __syncthreads();
  const float gmax = s_gmax;

  for (int chunk = 0; chunk < NCHUNK; ++chunk) {
    const int n0 = chunk * C_;
    const int cur = chunk & 1, nxt = cur ^ 1;
    const float* qsc = qs + cur * C_ * D_;
    const float* ksc = ks + cur * C_ * D_;
    const float* vsc = vs + cur * C_ * DH_;

    // ---- phase 1a: wait for this chunk's staged data ----
    wait_async_all();
    if (tid < C_) { sqk[tid] = 0.0f; zq[tid] = 0.0f; }
    __syncthreads();

    // ---- phase 1b: convert stage -> f16 operands; issue next chunk copies ----
    for (int e = tid; e < C_ * D_; e += 512) {
      float fq = qsc[e] * SCALE;
      float fk = ksc[e] * SCALE;
      xq[e] = (_Float16)fq;
      xk[e] = (_Float16)fk;
      atomicAdd(&sqk[e >> 6], 0.5f * fk * fk);
    }
    for (int e = tid; e < C_ * DH_; e += 512) {
      int i = e >> 5, d = e & (DH_ - 1);
      xvT[d * C_ + i] = (_Float16)vsc[e];
    }
    if (chunk + 1 < NCHUNK) {
      const float* qn = qb + (size_t)(n0 + C_) * D_;
      const float* kn = kb + (size_t)(n0 + C_) * D_;
      const float* vn = vb + (size_t)(n0 + C_) * D_;
      cp_async16(&qs[nxt * C_ * D_ + tid * 4], qn + tid * 4);
      cp_async16(&ks[nxt * C_ * D_ + tid * 4], kn + tid * 4);
      if (tid < 256) {
        int i = tid >> 3, seg = tid & 7;
        cp_async16(&vs[nxt * C_ * DH_ + i * DH_ + seg * 4], vn + i * D_ + dco + seg * 4);
      }
    }
    __syncthreads();

    // ---- phase 2: proj_q = xq @ omega^T ----
    proj_wmma<16>(xq, omh, proj);
    __syncthreads();

    // ---- phase 3: phi_q (per-row max cancels -0.5||x||^2) ----
    {
      int row = tid >> 4, part = tid & 15;      // 16 lanes per row
      float mx = -3.0e38f;
      for (int m = part; m < M_; m += 16) mx = fmaxf(mx, proj[row * M_ + m]);
      for (int off = 8; off > 0; off >>= 1) mx = fmaxf(mx, __shfl_xor(mx, off, 32));
      if (part == 0) rmax[row] = mx;
    }
    __syncthreads();
    for (int e = tid; e < C_ * M_; e += 512)
      pq[e] = (_Float16)(__expf(proj[e] - rmax[e >> 7]) * INV_SQRT_M + EPS_PHI);
    __syncthreads();

    // ---- phase 4: proj_k ----
    proj_wmma<16>(xk, omh, proj);
    __syncthreads();

    // ---- phase 5: phi_k (global max), row-major + transposed copies ----
    for (int e = tid; e < C_ * M_; e += 512) {
      int i = e >> 7, m = e & (M_ - 1);
      _Float16 h = (_Float16)(__expf(proj[e] - sqk[i] - gmax) * INV_SQRT_M + EPS_PHI);
      pk[e] = h;
      pkT[m * C_ + i] = h;
    }
    __syncthreads();

    // ---- phase 6: A = pq pk^T | O_inter = pq @ S_prev | zq = pq.z ----
    if (wave < 4) {                 // A: 2x2 tiles of [32,32], K=128
      int rt = wave >> 1, ct = wave & 1;
      v8f acc = {};
#pragma unroll
      for (int k0 = 0; k0 < M_; k0 += 32) {
        v16h a = load_a(pq, M_, rt * 16, k0);
        v16h b = load_bT(pk, M_, k0, ct * 16);   // pk rows ARE B^T layout
        acc = WMMA_F16(a, b, acc);
      }
      store_c(sa, C_, rt * 16, ct * 16, acc);
    } else if (wave < 8) {          // O_inter: 2x2 tiles of [32,32], K=128
      int t = wave - 4, rt = t >> 1, ct = t & 1;
      v8f acc = {};
#pragma unroll
      for (int k0 = 0; k0 < M_; k0 += 32) {
        v16h a = load_a(pq, M_, rt * 16, k0);
        v16h b = load_bT(ShT, M_, k0, ct * 16);  // ShT[d][m] contiguous
        acc = WMMA_F16(a, b, acc);
      }
      store_c(o, DH_, rt * 16, ct * 16, acc);
    } else {                        // zq_i = phi_q_i . z_prev (8 waves)
      int t = tid - 256;            // 0..255
      int i = t >> 3, part = t & 7;
      float s = 0.0f;
      for (int m = part; m < M_; m += 8) s += (float)pq[i * M_ + m] * z[m];
      atomicAdd(&zq[i], s);
    }
    __syncthreads();

    // ---- phase 7: causal mask + rowsum -> den ----
    if (tid < C_) {
      float rs = 0.0f;
      for (int j = 0; j < C_; ++j) {
        float av = (j <= tid) ? sa[tid * C_ + j] : 0.0f;
        am[tid * C_ + j] = (_Float16)av;
        rs += av;
      }
      den[tid] = zq[tid] + rs + EPS_DEN;
    }
    __syncthreads();

    // ---- phase 8: O += maskedA @ v | S += pk^T @ v | z += colsum(pk) ----
    if (wave < 4) {                 // O_intra: 2x2 tiles, K=32, C in place
      int rt = wave >> 1, ct = wave & 1;
      v8f c = load_c(o, DH_, rt * 16, ct * 16);
      v16h a = load_a(am, C_, rt * 16, 0);
      v16h b = load_bT(xvT, C_, 0, ct * 16);     // xvT[d][i] contiguous
      c = WMMA_F16(a, b, c);
      store_c(o, DH_, rt * 16, ct * 16, c);
    } else if (wave < 12) {         // S update: 16 tiles of [128,32], K=32
#pragma unroll
      for (int tt = 0; tt < 2; ++tt) {
        int idx = (wave - 4) * 2 + tt;
        int rt = idx >> 1, ct = idx & 1;
        v8f c = load_c(S, DH_, rt * 16, ct * 16);
        v16h a = load_a(pkT, C_, rt * 16, 0);    // pkT[m][i] contiguous
        v16h b = load_bT(xvT, C_, 0, ct * 16);
        c = WMMA_F16(a, b, c);
        store_c_dualT(S, ShT, DH_, rt * 16, ct * 16, c);
      }
    } else {                        // z += colsum(phi_k): 128 threads
      int m = tid - 384;
      float s = 0.0f;
      for (int i = 0; i < C_; ++i) s += (float)pkT[m * C_ + i];
      z[m] += s;
    }
    __syncthreads();

    // ---- phase 9: write out-half = o/den ----
    for (int e = tid; e < C_ * DH_; e += 512) {
      int i = e >> 5, d = e & (DH_ - 1);
      ob[(size_t)(n0 + i) * D_ + dco + d] = o[e] / den[i];
    }
    __syncthreads();
  }
}

// =========================== host launcher ===================================
extern "C" void kernel_launch(void* const* d_in, const int* in_sizes, int n_in,
                              void* d_out, int out_size, void* d_ws, size_t ws_size,
                              hipStream_t stream) {
  (void)in_sizes; (void)n_in; (void)out_size; (void)ws_size;
  const float* q  = (const float*)d_in[0];
  const float* k  = (const float*)d_in[1];
  const float* v  = (const float*)d_in[2];
  const float* om = (const float*)d_in[3];
  float* out = (float*)d_out;
  unsigned* ws = (unsigned*)d_ws;

  (void)hipMemsetAsync(d_ws, 0, sizeof(unsigned), stream);  // encoded -inf
  kmax_kernel<<<dim3(B_ * H_ * NCHUNK), dim3(256), 0, stream>>>(k, om, ws);
  performer_kernel<<<dim3(B_ * H_ * 2), dim3(512), 0, stream>>>(q, k, v, om, ws, out);
}
